// FermiLayer_57681410785847
// MI455X (gfx1250) — matrix operand order repaired and checked by reading, not compiled, not run on hardware
//
#include <hip/hip_runtime.h>

typedef __attribute__((ext_vector_type(16))) __bf16 v16bf;
typedef __attribute__((ext_vector_type(2)))  __bf16 v2bf;
typedef __attribute__((ext_vector_type(8)))  float  v8f;

#define RSQRT2 0.70710678118654752440f

__device__ __forceinline__ unsigned int pack_bf16(float a, float b) {
#if __has_builtin(__builtin_amdgcn_cvt_pk_bf16_f32)
    v2bf p = __builtin_amdgcn_cvt_pk_bf16_f32(a, b);
    return __builtin_bit_cast(unsigned int, p);
#else
    unsigned int ua = __float_as_uint(a);
    unsigned int ub = __float_as_uint(b);
    ua = (ua + (0x7FFFu + ((ua >> 16) & 1u))) >> 16;
    ub = (ub + (0x7FFFu + ((ub >> 16) & 1u))) >> 16;
    return (ua & 0xFFFFu) | ((ub & 0xFFFFu) << 16);
#endif
}

// tanh: prefer the gfx1250 hardware v_tanh_f32 (single TRANS32 op, co-executes
// with WMMA).  Fallback: 1 - 2*rcp(e^{2x}+1) via raw v_exp_f32 / v_rcp_f32 —
// branch-free, saturates correctly at +/-1.
__device__ __forceinline__ float fast_tanh(float x) {
#if __has_builtin(__builtin_amdgcn_tanhf)
    return __builtin_amdgcn_tanhf(x);
#else
    float e = __expf(2.0f * x);
    return 1.0f - 2.0f * __builtin_amdgcn_rcpf(e + 1.0f);
#endif
}

// ---------------------------------------------------------------------------
// Kernel 0: pack W_same / W_diff into bf16 WMMA B-fragment layout; zero gsum.
// Fragment layout (B of v_wmma_f32_16x16x32_bf16, 32x16 K-major):
//   lane L: column n = n0*16 + (L&15), K-half kh = L>>4
//   dword v (0..7): bf16 pair (K, K+1) with K = k0*32 + kh*16 + 2*v
// Stored so the pair kernel loads dword v at [frag*256 + v*32 + lane].
// ---------------------------------------------------------------------------
__global__ __launch_bounds__(256) void prep_kernel(const float* __restrict__ Wsame,
                                                   const float* __restrict__ Wdiff,
                                                   unsigned int* __restrict__ wpack,
                                                   float* __restrict__ gsum) {
    int gtid = blockIdx.x * 256 + threadIdx.x;
    if (gtid < 131072) gsum[gtid] = 0.0f;           // [1024][2][64] spin sums
    if (gtid < 4096) {
        int fid  = gtid >> 8;
        int rem  = gtid & 255;
        int v    = rem >> 5;
        int lane = rem & 31;
        int w    = fid >> 3;
        int k0   = (fid >> 2) & 1;
        int n0   = fid & 3;
        int kh   = lane >> 4;
        int n    = n0 * 16 + (lane & 15);
        int K    = k0 * 32 + kh * 16 + 2 * v;
        const float* W = w ? Wdiff : Wsame;          // row-major [64][64]
        wpack[gtid] = pack_bf16(W[K * 64 + n], W[(K + 1) * 64 + n]);
    }
}

// ---------------------------------------------------------------------------
// Kernel 1: pair stream.  Block b: i = b>>3, j in [(b&7)*128, +128) — one spin
// block, one weight matrix.  Wave handles 16 rows x 64 cols:
//   2 A-frags (K 0..31 / 32..63) x 4 column tiles x 2 K-steps = 8 WMMA.
// Fused: per-(i,half,d) spin sums of h_two via shfl + LDS + global f32 atomic.
// ---------------------------------------------------------------------------
__global__ __launch_bounds__(256) void pair_kernel(const float* __restrict__ h_two,
                                                   const unsigned int* __restrict__ wpack,
                                                   const float* __restrict__ b_same,
                                                   const float* __restrict__ b_diff,
                                                   float* __restrict__ out2,
                                                   float* __restrict__ gsum) {
    __shared__ float tile[8][16 * 72];   // pitch 72 floats: 16B-aligned rows
    __shared__ float wsum[8][64];

    const int lane = threadIdx.x & 31;
    const int wave = threadIdx.x >> 5;
    const int b    = blockIdx.x;
    const int i    = b >> 3;
    const int jhalf = ((b & 7) >= 4) ? 1 : 0;
    const int w    = ((i < 512) == (jhalf == 0)) ? 0 : 1;   // 0 = same, 1 = diff
    const float* bias = w ? b_diff : b_same;

    const size_t rowBase = (size_t)b * 128 + (size_t)wave * 16;

    // ---- stage 16x64 f32 tile into LDS (coalesced b128 global loads) ----
    const float4* src = (const float4*)(h_two + rowBase * 64);
    float* t = tile[wave];
    #pragma unroll
    for (int c = 0; c < 8; ++c) {
        int idx = c * 32 + lane;
        float4 d = src[idx];
        int r  = idx >> 4;
        int c4 = idx & 15;
        *(float4*)&t[r * 72 + c4 * 4] = d;
    }
    __builtin_amdgcn_wave_barrier();
    asm volatile("s_wait_dscnt 0" ::: "memory");   // wave-private tile: dscnt only

    // ---- build A fragments (16x32 bf16, contiguous K = hi*16 + 2v, same order as wpack B) ----
    const int hi = lane >> 4;
    const int r  = lane & 15;
    union { unsigned int u[8]; v16bf v; } a0, a1;
    #pragma unroll
    for (int v = 0; v < 8; ++v) {
        int K = hi * 16 + 2 * v;
        float2 p0 = *(const float2*)&t[r * 72 + K];
        float2 p1 = *(const float2*)&t[r * 72 + 32 + K];
        a0.u[v] = pack_bf16(p0.x, p0.y);
        a1.u[v] = pack_bf16(p1.x, p1.y);
    }

    const unsigned int* wp = wpack + (size_t)w * 2048;
    #pragma unroll
    for (int n0 = 0; n0 < 4; ++n0) {
        union { unsigned int u[8]; v16bf v; } b0, b1;
        #pragma unroll
        for (int vv = 0; vv < 8; ++vv) {
            b0.u[vv] = wp[(0 * 4 + n0) * 256 + vv * 32 + lane];
            b1.u[vv] = wp[(1 * 4 + n0) * 256 + vv * 32 + lane];
        }
        v8f acc = {};
        acc = __builtin_amdgcn_wmma_f32_16x16x32_bf16(false, a0.v, false, b0.v,
                                                      (short)0, acc, false, false);
        acc = __builtin_amdgcn_wmma_f32_16x16x32_bf16(false, a1.v, false, b1.v,
                                                      (short)0, acc, false, false);

        // ---- epilogue: bias + tanh + residual + fused spin-sum ----
        const int n = n0 * 16 + r;
        const float bv = bias[n];
        float psum = 0.0f;
        #pragma unroll
        for (int vv = 0; vv < 8; ++vv) {
            int m   = hi * 8 + vv;                 // D layout: lane hi picks row half
            float x = t[m * 72 + n];               // original h_two value
            float y = acc[vv] + bv;
            out2[(rowBase + (size_t)m) * 64 + n] = (x + fast_tanh(y)) * RSQRT2;
            psum += x;
        }
        psum += __shfl_xor(psum, 16);              // combine the two row-halves
        if (hi == 0) wsum[wave][n] = psum;         // 16-row sum for feature n
    }

    __syncthreads();
    if (threadIdx.x < 64) {
        float s = 0.0f;
        #pragma unroll
        for (int ww = 0; ww < 8; ++ww) s += wsum[ww][threadIdx.x];
        unsafeAtomicAdd(&gsum[(size_t)i * 128 + jhalf * 64 + threadIdx.x], s);
    }
}

// ---------------------------------------------------------------------------
// Kernel 2: g_up/g_dn means of h_one halves, then g_new = g_in @ W_g  [2,256]
// ---------------------------------------------------------------------------
__global__ __launch_bounds__(256) void gnew_kernel(const float* __restrict__ h_one,
                                                   const float* __restrict__ W_g,
                                                   float* __restrict__ gnew) {
    __shared__ float sgu[256], sgd[256];
    const int tdx = threadIdx.x;
    float su = 0.0f, sd = 0.0f;
    for (int ii = 0;   ii < 512;  ++ii) su += h_one[ii * 256 + tdx];
    for (int ii = 512; ii < 1024; ++ii) sd += h_one[ii * 256 + tdx];
    sgu[tdx] = su * (1.0f / 512.0f);
    sgd[tdx] = sd * (1.0f / 512.0f);
    __syncthreads();
    float a0 = 0.0f, a1 = 0.0f;
    for (int k = 0; k < 256; ++k) {
        float w0 = W_g[k * 256 + tdx];
        float w1 = W_g[(256 + k) * 256 + tdx];
        a0 += sgu[k] * w0 + sgd[k] * w1;   // row [g_up, g_dn]
        a1 += sgd[k] * w0 + sgu[k] * w1;   // row [g_dn, g_up]
    }
    gnew[tdx]       = a0;
    gnew[256 + tdx] = a1;
}

// ---------------------------------------------------------------------------
// Kernel 3: single stream.  Block i: one_in = [h_one[i], pf(64), pf(64)];
// h_out = (h_one + tanh((one_in@W_one + b_one + g_new)/sqrt2)) / sqrt2
// ---------------------------------------------------------------------------
__global__ __launch_bounds__(256) void single_kernel(const float* __restrict__ h_one,
                                                     const float* __restrict__ W_one,
                                                     const float* __restrict__ b_one,
                                                     const float* __restrict__ gsum,
                                                     const float* __restrict__ gnew,
                                                     float* __restrict__ out1) {
    __shared__ float onein[384];
    const int i  = blockIdx.x;
    const int tdx = threadIdx.x;
    const int up = (i < 512) ? 1 : 0;
    onein[tdx] = h_one[i * 256 + tdx];
    if (tdx < 128) {
        int slot = tdx >> 6;                 // pf slot 0 / 1
        int d    = tdx & 63;
        int hsel = up ? slot : (1 - slot);   // down electrons: channels swapped
        onein[256 + tdx] = gsum[(size_t)i * 128 + hsel * 64 + d] * (1.0f / 512.0f);
    }
    __syncthreads();
    float acc = b_one[tdx] + gnew[(up ? 0 : 1) * 256 + tdx];
    for (int k = 0; k < 384; ++k) acc += onein[k] * W_one[k * 256 + tdx];
    float tt = fast_tanh(acc * RSQRT2);
    out1[i * 256 + tdx] = (h_one[i * 256 + tdx] + tt) * RSQRT2;
}

// ---------------------------------------------------------------------------
extern "C" void kernel_launch(void* const* d_in, const int* in_sizes, int n_in,
                              void* d_out, int out_size, void* d_ws, size_t ws_size,
                              hipStream_t stream) {
    const float* h_one  = (const float*)d_in[0];
    const float* h_two  = (const float*)d_in[1];
    const float* W_one  = (const float*)d_in[2];
    const float* b_one  = (const float*)d_in[3];
    const float* W_g    = (const float*)d_in[4];
    const float* W_same = (const float*)d_in[5];
    const float* b_same = (const float*)d_in[6];
    const float* W_diff = (const float*)d_in[7];
    const float* b_diff = (const float*)d_in[8];

    float* out1 = (float*)d_out;                 // h_one_out [1024,256]
    float* out2 = out1 + 1024 * 256;             // h_two_out [1024,1024,64]

    unsigned int* wpack = (unsigned int*)d_ws;   // 4096 dwords bf16 W frags
    float* gsum = (float*)d_ws + 4096;           // [1024][2][64] spin sums
    float* gnew = gsum + 131072;                 // [2][256]

    prep_kernel  <<<512,  256, 0, stream>>>(W_same, W_diff, wpack, gsum);
    pair_kernel  <<<8192, 256, 0, stream>>>(h_two, wpack, b_same, b_diff, out2, gsum);
    gnew_kernel  <<<1,    256, 0, stream>>>(h_one, W_g, gnew);
    single_kernel<<<1024, 256, 0, stream>>>(h_one, W_one, b_one, gsum, gnew, out1);
}
